// UnGraphSAINT_47115791237273
// MI455X (gfx1250) — matrix operand order repaired and checked by reading, not compiled, not run on hardware
//
#include <hip/hip_runtime.h>
#include <hip/hip_bf16.h>

// ---------------------------------------------------------------------------
// GraphSAINT 2-layer HighOrderAggregator forward for MI455X (gfx1250).
// GEMMs run on WMMA bf16 matrix cores (fp32 accumulate) with a software-
// pipelined fragment loop (prefetch next k-step before issuing WMMAs so the
// loadcnt waits are partial, not full). SpMM uses global_atomic_add_f32.
// ---------------------------------------------------------------------------

typedef unsigned short u16;
typedef unsigned int   u32;

typedef __attribute__((ext_vector_type(16))) __bf16 v16bf;
typedef __attribute__((ext_vector_type(8)))  float  v8f;

union FragBF { v16bf v; uint4 q[2]; };

static constexpr int NSUB = 32768;
static constexpr int FIN  = 512;
static constexpr int NNZE = 1048576;
static constexpr int NOUT = 512;      // every GEMM in this net has N = 512

__device__ __forceinline__ u16 f2bf(float x) {
    union { float f; u32 u; } v; v.f = x;
    u32 r = v.u + 0x7FFFu + ((v.u >> 16) & 1u);   // round-to-nearest-even
    return (u16)(r >> 16);
}

// -------------------------- gather: feat_full[idx] --------------------------
__global__ __launch_bounds__(256) void gather_kernel(
    const float* __restrict__ feat_full, const int* __restrict__ idx,
    float* __restrict__ out) {
    int t  = blockIdx.x * 256 + threadIdx.x;      // over NSUB * (FIN/4)
    int i  = t >> 7;                              // FIN/4 = 128
    int j4 = t & 127;
    const float4* src = (const float4*)(feat_full + (size_t)idx[i] * FIN);
    ((float4*)(out + (size_t)i * FIN))[j4] = src[j4];
}

// -------------------------- zero fill (float4) ------------------------------
__global__ __launch_bounds__(256) void zero_kernel(float4* __restrict__ p, int n4) {
    int t = blockIdx.x * 256 + threadIdx.x;
    if (t < n4) p[t] = make_float4(0.f, 0.f, 0.f, 0.f);
}

// -------------------------- f32 -> bf16 convert -----------------------------
__global__ __launch_bounds__(256) void convert_bf16_kernel(
    const float* __restrict__ in, u16* __restrict__ out, int n) {
    int t = blockIdx.x * 256 + threadIdx.x;
    if (t < n) out[t] = f2bf(in[t]);
}

// -------------------------- SpMM scatter: out[row] += val * feat[col] -------
__global__ __launch_bounds__(256) void spmm_scatter_kernel(
    const int* __restrict__ arow, const int* __restrict__ acol,
    const float* __restrict__ aval, const float* __restrict__ feat,
    float* __restrict__ out, int D) {
    int e = blockIdx.x;
    int r = arow[e], c = acol[e];
    float v = aval[e];
    const float4* f4 = (const float4*)(feat + (size_t)c * D);
    float* orow = out + (size_t)r * D;
    for (int j = threadIdx.x; j < (D >> 2); j += blockDim.x) {
        float4 x = f4[j];
        atomicAdd(orow + 4 * j + 0, v * x.x);
        atomicAdd(orow + 4 * j + 1, v * x.y);
        atomicAdd(orow + 4 * j + 2, v * x.z);
        atomicAdd(orow + 4 * j + 3, v * x.w);
    }
}

// -------------------------- WMMA bf16 GEMM + fused epilogue -----------------
// C[m,n] = sum_k A[m,k] * W[n,k]   (A: [M,K] bf16 row-major, W: [512,K] bf16)
// NORM=1: f = relu(C + b); per-row mean/var norm; out = (f-mean)*s*rstd + o
// NORM=0: out = C + b
// Block = 256 threads (8 waves), computes a 16 x 512 output slab.
template <bool NORM>
__global__ __launch_bounds__(256) void gemm_wmma_kernel(
    const u16* __restrict__ Abf, const u16* __restrict__ Wbf,
    const float* __restrict__ bias, const float* __restrict__ scale,
    const float* __restrict__ offs, float* __restrict__ out,
    int K, int ldout, int col_off) {
    __shared__ float slab[16 * 516];              // padded: bank = (4m+n)%64
    __shared__ float redS[256], redSS[256];
    __shared__ float mean_s[16], rstd_s[16];

    const int lane   = threadIdx.x & 31;          // wave32
    const int wave   = threadIdx.x >> 5;
    const int lanelo = lane & 15;
    const int half   = lane >> 4;                 // 0: lanes 0-15, 1: 16-31
    const int mblk   = blockIdx.x * 16;

    // A 16x32 bf16 fragment (ISA 7.12.2): lane half 0 holds K base 0..7 and
    // 16..23; half 1 holds 8..15 and 24..31.  Row m = lanelo.
    const int abase = half * 8;
    const u16* arow = Abf + (size_t)(mblk + lanelo) * K + abase;

    // B 32x16 bf16 fragment: same K order as A (half 0 holds K 0..7 and
    // 16..23; half 1 holds 8..15 and 24..31).  Col n = lanelo.
    const int bbase = half * 8;
    const u16* brow[4];
#pragma unroll
    for (int t = 0; t < 4; ++t) {
        int n = (wave * 4 + t) * 16 + lanelo;
        brow[t] = Wbf + (size_t)n * K + bbase;
    }

    // Software-pipelined K loop: fragments for step k are loaded in the
    // previous iteration, so the wait before the WMMA group only has to cover
    // the (still in-flight) next-step loads -> partial s_wait_loadcnt.
    v8f acc[4] = {};
    FragBF a_c, b_c[4];
    a_c.q[0] = *(const uint4*)(arow);
    a_c.q[1] = *(const uint4*)(arow + 16);
#pragma unroll
    for (int t = 0; t < 4; ++t) {
        b_c[t].q[0] = *(const uint4*)(brow[t]);
        b_c[t].q[1] = *(const uint4*)(brow[t] + 16);
    }
    for (int k = 0; k < K; k += 32) {
        const int kn = (k + 32 < K) ? (k + 32) : 0;   // wrap: dummy, in-bounds
        FragBF a_n, b_n[4];
        a_n.q[0] = *(const uint4*)(arow + kn);
        a_n.q[1] = *(const uint4*)(arow + kn + 16);
#pragma unroll
        for (int t = 0; t < 4; ++t) {
            b_n[t].q[0] = *(const uint4*)(brow[t] + kn);
            b_n[t].q[1] = *(const uint4*)(brow[t] + kn + 16);
        }
#pragma unroll
        for (int t = 0; t < 4; ++t)
            acc[t] = __builtin_amdgcn_wmma_f32_16x16x32_bf16(
                false, a_c.v, false, b_c[t].v, (short)0, acc[t], false, false);
        a_c = a_n;
#pragma unroll
        for (int t = 0; t < 4; ++t) b_c[t] = b_n[t];
    }

    // D/C layout: lane l holds col n = l&15; VGPR v holds row m = v + 8*half.
    if (NORM) {
#pragma unroll
        for (int t = 0; t < 4; ++t) {
            int n = (wave * 4 + t) * 16 + lanelo;
            float bi = bias[n];
#pragma unroll
            for (int v = 0; v < 8; ++v) {
                int m = half * 8 + v;
                float f = acc[t][v] + bi;
                slab[m * 516 + n] = fmaxf(f, 0.f);
            }
        }
        __syncthreads();
        // mean/var: 16 threads per row, 32 elements each, then 16-way combine
        {
            int r = threadIdx.x >> 4;
            int c = threadIdx.x & 15;
            const float* sr = slab + r * 516;
            float s = 0.f, ss = 0.f;
#pragma unroll
            for (int j = c; j < NOUT; j += 16) {
                float x = sr[j]; s += x; ss += x * x;
            }
            redS[threadIdx.x] = s;
            redSS[threadIdx.x] = ss;
        }
        __syncthreads();
        if (threadIdx.x < 16) {
            int r = threadIdx.x;
            float s = 0.f, ss = 0.f;
#pragma unroll
            for (int c = 0; c < 16; ++c) { s += redS[r * 16 + c]; ss += redSS[r * 16 + c]; }
            float mean = s * (1.f / NOUT);
            float var  = ss * (1.f / NOUT) - mean * mean;
            mean_s[r] = mean;
            rstd_s[r] = rsqrtf(var + 1e-9f);
        }
        __syncthreads();
        for (int i = threadIdx.x; i < 16 * NOUT; i += 256) {
            int m = i >> 9, n = i & 511;
            float f = slab[m * 516 + n];
            out[(size_t)(mblk + m) * ldout + col_off + n] =
                (f - mean_s[m]) * scale[n] * rstd_s[m] + offs[n];
        }
    } else {
#pragma unroll
        for (int t = 0; t < 4; ++t) {
            int n = (wave * 4 + t) * 16 + lanelo;
            float bi = bias[n];
#pragma unroll
            for (int v = 0; v < 8; ++v) {
                int m = half * 8 + v;
                out[(size_t)(mblk + m) * ldout + col_off + n] = acc[t][v] + bi;
            }
        }
    }
}

// -------------------------- row L2 normalize -> bf16 ------------------------
__global__ __launch_bounds__(256) void l2norm_bf16_kernel(
    const float* __restrict__ in, u16* __restrict__ out, int D) {
    __shared__ float red[256];
    size_t row = blockIdx.x;
    const float* r = in + row * D;
    float s = 0.f;
    for (int j = threadIdx.x; j < D; j += 256) { float x = r[j]; s += x * x; }
    red[threadIdx.x] = s;
    __syncthreads();
    for (int off = 128; off > 0; off >>= 1) {
        if (threadIdx.x < (unsigned)off) red[threadIdx.x] += red[threadIdx.x + off];
        __syncthreads();
    }
    float inv = 1.f / fmaxf(sqrtf(red[0]), 1e-12f);
    u16* o = out + row * D;
    for (int j = threadIdx.x; j < D; j += 256) o[j] = f2bf(r[j] * inv);
}

// ---------------------------------------------------------------------------
extern "C" void kernel_launch(void* const* d_in, const int* in_sizes, int n_in,
                              void* d_out, int out_size, void* d_ws, size_t ws_size,
                              hipStream_t stream) {
    (void)in_sizes; (void)n_in; (void)out_size; (void)ws_size;

    const float* feat_full = (const float*)d_in[0];
    const int*   node_sub  = (const int*)d_in[1];
    const int*   adj_row   = (const int*)d_in[2];
    const int*   adj_col   = (const int*)d_in[3];
    const float* adj_val   = (const float*)d_in[4];
    const float* W1_0 = (const float*)d_in[5],  *W1_1 = (const float*)d_in[6];
    const float* b1_0 = (const float*)d_in[7],  *b1_1 = (const float*)d_in[8];
    const float* s1_0 = (const float*)d_in[9],  *s1_1 = (const float*)d_in[10];
    const float* o1_0 = (const float*)d_in[11], *o1_1 = (const float*)d_in[12];
    const float* W2_0 = (const float*)d_in[13], *W2_1 = (const float*)d_in[14];
    const float* b2_0 = (const float*)d_in[15], *b2_1 = (const float*)d_in[16];
    const float* s2_0 = (const float*)d_in[17], *s2_1 = (const float*)d_in[18];
    const float* o2_0 = (const float*)d_in[19], *o2_1 = (const float*)d_in[20];
    const float* Wc   = (const float*)d_in[21];
    const float* bc   = (const float*)d_in[22];

    // Workspace layout (bytes). G (layer-2 out, 128MB) overlays A+B which are
    // dead by the time G is written.
    const size_t MB = 1u << 20;
    char* ws = (char*)d_ws;
    float* A = (float*)(ws + 0);          // [NSUB,512]  gathered feat
    float* B = (float*)(ws + 64 * MB);    // [NSUB,512]  spmm1 out
    float* G = (float*)(ws + 0);          // [NSUB,1024] layer2 out (overlay)
    float* C = (float*)(ws + 128 * MB);   // [NSUB,1024] layer1 out
    float* D = (float*)(ws + 256 * MB);   // [NSUB,1024] spmm2 out
    u16*   E = (u16*)  (ws + 384 * MB);   // [NSUB,1024] bf16 GEMM-A scratch
    u16* W1_0b = (u16*)(ws + 448 * MB);                    // 512*512
    u16* W1_1b = (u16*)(ws + 448 * MB + 512 * 1024);       // 512*512
    u16* W2_0b = (u16*)(ws + 449 * MB);                    // 512*1024
    u16* W2_1b = (u16*)(ws + 450 * MB);                    // 512*1024
    u16* Wcb   = (u16*)(ws + 451 * MB);                    // 512*1024

    const int TB = 256;
    auto cvt = [&](const float* src, u16* dst, int n) {
        convert_bf16_kernel<<<(n + TB - 1) / TB, TB, 0, stream>>>(src, dst, n);
    };

    // Weights -> bf16 (once per call; deterministic).
    cvt(W1_0, W1_0b, 512 * 512);
    cvt(W1_1, W1_1b, 512 * 512);
    cvt(W2_0, W2_0b, 512 * 1024);
    cvt(W2_1, W2_1b, 512 * 1024);
    cvt(Wc,   Wcb,   512 * 1024);

    // 1) gather
    gather_kernel<<<(NSUB * (FIN / 4)) / TB, TB, 0, stream>>>(feat_full, node_sub, A);

    // 2) spmm1: B = Adj @ A   (zero first; ws is poisoned)
    {
        int n4 = NSUB * FIN / 4;
        zero_kernel<<<(n4 + TB - 1) / TB, TB, 0, stream>>>((float4*)B, n4);
        spmm_scatter_kernel<<<NNZE, 128, 0, stream>>>(adj_row, adj_col, adj_val, A, B, FIN);
    }

    // 3) layer-1 part 0: C[:, 0:512] from A
    cvt(A, E, NSUB * 512);
    gemm_wmma_kernel<true><<<NSUB / 16, TB, 0, stream>>>(E, W1_0b, b1_0, s1_0, o1_0,
                                                         C, 512, 1024, 0);
    // 4) layer-1 part 1: C[:, 512:1024] from B
    cvt(B, E, NSUB * 512);
    gemm_wmma_kernel<true><<<NSUB / 16, TB, 0, stream>>>(E, W1_1b, b1_1, s1_1, o1_1,
                                                         C, 512, 1024, 512);

    // 5) spmm2: D = Adj @ C
    {
        int n4 = NSUB * 1024 / 4;
        zero_kernel<<<(n4 + TB - 1) / TB, TB, 0, stream>>>((float4*)D, n4);
        spmm_scatter_kernel<<<NNZE, 256, 0, stream>>>(adj_row, adj_col, adj_val, C, D, 1024);
    }

    // 6) layer-2 part 0: G[:, 0:512] from C   (A/B dead -> G overlay safe)
    cvt(C, E, NSUB * 1024);
    gemm_wmma_kernel<true><<<NSUB / 16, TB, 0, stream>>>(E, W2_0b, b2_0, s2_0, o2_0,
                                                         G, 1024, 1024, 0);
    // 7) layer-2 part 1: G[:, 512:1024] from D
    cvt(D, E, NSUB * 1024);
    gemm_wmma_kernel<true><<<NSUB / 16, TB, 0, stream>>>(E, W2_1b, b2_1, s2_1, o2_1,
                                                         G, 1024, 1024, 512);

    // 8) row L2 normalize -> bf16
    l2norm_bf16_kernel<<<NSUB, TB, 0, stream>>>(G, E, 1024);

    // 9) classifier: out = norm(h) @ Wc^T + bc
    gemm_wmma_kernel<false><<<NSUB / 16, TB, 0, stream>>>(E, Wcb, bc, nullptr, nullptr,
                                                          (float*)d_out, 1024, 512, 0);
}
